// Attention_68882685494025
// MI455X (gfx1250) — matrix operand order repaired and checked by reading, not compiled, not run on hardware
//
#include <hip/hip_runtime.h>

// ---------------------------------------------------------------------------
// BEiT window attention for MI455X (gfx1250), wave32 + WMMA f16->f32.
//   B=64, N=197 (padded 224), C=768, H=12, D=64
//   kernel 0: f32 -> f16 conversion passes (x, qkv_w, proj_w)   [once]
//   kernel 1: qkv  = xh @ qkv_wh^T + bias        (f16 in -> f16 out, 32x64/wave)
//   kernel 2: per (b,h): scores WMMA -> +relpos -> softmax -> @V WMMA (LDS),
//             Q/K staged with global_load_async_to_lds_b128 (ASYNCcnt)
//   kernel 3: out  = ctx @ proj_wh^T + proj_b    (f16 in -> f32 out, 32x64/wave)
// ---------------------------------------------------------------------------

typedef __attribute__((ext_vector_type(16))) _Float16 v16h;
typedef __attribute__((ext_vector_type(8)))  _Float16 h8;
typedef __attribute__((ext_vector_type(8)))  float    v8f;
typedef __attribute__((ext_vector_type(4)))  float    f4v;

#define WMMA_F16(a, b, c) \
  __builtin_amdgcn_wmma_f32_16x16x32_f16(false, (a), false, (b), (short)0, (c), false, false)

// A-fragment (16x32 f16) from a row-major f16 row: two 8-half chunks at +0,+8.
static __device__ __forceinline__ v16h fragA_from_h(const _Float16* p) {
  h8 lo = *(const h8*)(p);
  h8 hi = *(const h8*)(p + 8);
  return __builtin_shufflevector(lo, hi, 0, 1, 2, 3, 4, 5, 6, 7,
                                 8, 9, 10, 11, 12, 13, 14, 15);
}

// A-fragment from row-major f32 (for softmax probabilities in LDS).
static __device__ __forceinline__ v16h fragA_from_f32(const float* p) {
  f4v x0 = *(const f4v*)(p);
  f4v x1 = *(const f4v*)(p + 4);
  f4v x2 = *(const f4v*)(p + 8);
  f4v x3 = *(const f4v*)(p + 12);
  v16h a;
#pragma unroll
  for (int i = 0; i < 4; ++i) {
    a[i]      = (_Float16)x0[i];
    a[i + 4]  = (_Float16)x1[i];
    a[i + 8]  = (_Float16)x2[i];
    a[i + 12] = (_Float16)x3[i];
  }
  return a;
}

// ---------------------------------------------------------------------------
// Kernel 0: elementwise f32 -> f16 (n divisible by 4).
// ---------------------------------------------------------------------------
__global__ __launch_bounds__(256) void cvt_f16_kernel(
    const float* __restrict__ src, _Float16* __restrict__ dst, int n) {
  const int i = (blockIdx.x * 256 + threadIdx.x) * 4;
  if (i < n) {
    f4v x = *(const f4v*)(src + i);
#pragma unroll
    for (int j = 0; j < 4; ++j) dst[i + j] = (_Float16)x[j];
  }
}

// ---------------------------------------------------------------------------
// Kernel 1: qkv GEMM. out(12608, 2304) f16 = Xh(12608,768) @ Wh(2304,768)^T + b
// One wave -> 32x64 tile: 2 A-frags x 4 B-frags, 8 accumulators.
// ---------------------------------------------------------------------------
__global__ __launch_bounds__(256) void qkv_gemm_kernel(
    const _Float16* __restrict__ Xh, const _Float16* __restrict__ Wh,
    const float* __restrict__ qb, const float* __restrict__ kb,
    const float* __restrict__ vb, _Float16* __restrict__ out) {
  constexpr int K = 768, N = 2304, NT = 36;  // N / 64
  const int lane = threadIdx.x & 31;
  const int wv = blockIdx.x * 8 + (threadIdx.x >> 5);
  const int mt = wv / NT, nt = wv - mt * NT;
  const int m0 = mt * 32, n0 = nt * 64;

  const _Float16* ap0 = Xh + (size_t)(m0 + (lane & 15)) * K + ((lane >> 4) << 4);
  const _Float16* ap1 = ap0 + (size_t)16 * K;
  const _Float16* bp  = Wh + (size_t)(n0 + (lane & 15)) * K + ((lane >> 4) << 4);

  v8f acc[2][4] = {};
  for (int kc = 0; kc < K; kc += 32) {
    __builtin_prefetch(ap0 + kc + 256, 0, 1);
    __builtin_prefetch(bp + kc + 256, 0, 1);
    v16h a0 = fragA_from_h(ap0 + kc);
    v16h a1 = fragA_from_h(ap1 + kc);
#pragma unroll
    for (int s = 0; s < 4; ++s) {
      v16h b = *(const v16h*)(bp + (size_t)s * 16 * K + kc);
      acc[0][s] = WMMA_F16(a0, b, acc[0][s]);
      acc[1][s] = WMMA_F16(a1, b, acc[1][s]);
    }
  }

  const int r8 = (lane >> 4) << 3;
#pragma unroll
  for (int s = 0; s < 4; ++s) {
    const int col = n0 + s * 16 + (lane & 15);
    const float bias =
        (col < 768) ? qb[col] : (col < 1536 ? kb[col - 768] : vb[col - 1536]);
#pragma unroll
    for (int half = 0; half < 2; ++half) {
#pragma unroll
      for (int r = 0; r < 8; ++r) {
        const int row = m0 + half * 16 + r + r8;
        out[(size_t)row * N + col] = (_Float16)(acc[half][s][r] + bias);
      }
    }
  }
}

// ---------------------------------------------------------------------------
// Kernel 2: attention for one (b, h) per workgroup. N padded 197 -> 224.
// LDS: f32 scores 224x224 | f16 Q 224x64 | f16 K 224x64 | f16 V^T 64x224 |
//      f32 rel column 732    (total 289,648 B < 320 KB/WGP)
// ---------------------------------------------------------------------------
#define NPAD 224

__global__ __launch_bounds__(256) void attn_kernel(
    const _Float16* __restrict__ qkvh, const float* __restrict__ rel_table,
    _Float16* __restrict__ ctx) {
  extern __shared__ char smem[];
  float*    s_sc  = (float*)smem;                            // 224*224 f32
  _Float16* s_q   = (_Float16*)(smem + NPAD * NPAD * 4);     // 224*64
  _Float16* s_k   = s_q + NPAD * 64;                         // 224*64
  _Float16* s_vt  = s_k + NPAD * 64;                         // 64*224 (transposed)
  float*    s_rel = (float*)(s_vt + 64 * NPAD);              // 732

  const int b = blockIdx.x / 12, h = blockIdx.x % 12;
  const int tid = threadIdx.x, lane = tid & 31, wid = tid >> 5;
  const size_t base = (size_t)b * 197 * 2304;
  const int hoff = h * 64;

  // ---- phase 0a: zero Q/K pad rows (197..223) and stage V^T manually ----
  for (int t = tid; t < 27 * 64; t += 256) {
    s_q[197 * 64 + t] = (_Float16)0;
    s_k[197 * 64 + t] = (_Float16)0;
  }
  for (int t = tid; t < NPAD * 64; t += 256) {
    const int row = t >> 6, d = t & 63;
    _Float16 vv = (_Float16)0;
    if (row < 197) vv = qkvh[base + (size_t)row * 2304 + 1536 + hoff + d];
    s_vt[d * NPAD + row] = vv;
  }
  for (int t = tid; t < 732; t += 256) s_rel[t] = rel_table[t * 12 + h];

  // ---- phase 0b: async-copy Q and K rows into LDS (CDNA5 ASYNC path) ----
  // 197 rows x 128 B = 1576 16-byte chunks per matrix.
  {
    const unsigned q_lds = (unsigned)(size_t)s_q;
    const unsigned k_lds = (unsigned)(size_t)s_k;
    for (int t = tid; t < 1576; t += 256) {
      const int row = t >> 3, c = t & 7;  // chunk c covers halves 8c..8c+7
      const _Float16* gq = qkvh + base + (size_t)row * 2304 + hoff + c * 8;
      const _Float16* gk = gq + 768;
      const unsigned lq = q_lds + (unsigned)t * 16u;
      const unsigned lk = k_lds + (unsigned)t * 16u;
      asm volatile("global_load_async_to_lds_b128 %0, %1, off"
                   :: "v"(lq), "v"(gq) : "memory");
      asm volatile("global_load_async_to_lds_b128 %0, %1, off"
                   :: "v"(lk), "v"(gk) : "memory");
    }
    asm volatile("s_wait_asynccnt 0x0" ::: "memory");
  }
  __syncthreads();

  const int koffA = (lane >> 4) << 4;   // 0 or 16
  const int koffB = (lane >> 4) << 4;   // 0 or 16
  const int r8    = (lane >> 4) << 3;

  // ---- phase 1: scores = Q @ K^T  (14x14 tiles of 16x16, K-dim = 64) ----
  for (int t = wid; t < 196; t += 8) {
    const int ti = t / 14, tj = t - ti * 14;
    const int m0 = ti * 16, n0 = tj * 16;
    const _Float16* ap = s_q + (m0 + (lane & 15)) * 64 + koffA;
    const _Float16* bp = s_k + (n0 + (lane & 15)) * 64 + koffB;
    v8f acc = {};
#pragma unroll
    for (int kc = 0; kc < 64; kc += 32) {
      v16h a  = fragA_from_h(ap + kc);
      v16h bf = *(const v16h*)(bp + kc);
      acc = WMMA_F16(a, bf, acc);
    }
#pragma unroll
    for (int r = 0; r < 8; ++r)
      s_sc[(m0 + r + r8) * NPAD + n0 + (lane & 15)] = acc[r];
  }
  __syncthreads();

  // ---- phase 2: +relative-position bias, softmax (one thread per row) ----
  if (tid < NPAD) {
    const int row = tid;
    float* sr = s_sc + row * NPAD;
    if (row < 197) {
      const float scale = 0.125f;  // 64^-0.5
      const int p = row - 1;
      const int ph = (p >= 0) ? p / 14 : 0;
      const int pw = (p >= 0) ? p - ph * 14 : 0;
      float mx = -1e30f;
      for (int m = 0; m < 197; ++m) {
        int idx;
        if (row == 0 && m == 0)      idx = 731;      // nrd-1
        else if (m == 0)             idx = 730;      // nrd-2
        else if (row == 0)           idx = 729;      // nrd-3
        else {
          const int q = m - 1;
          const int qh = q / 14, qw = q - qh * 14;
          idx = (ph - qh + 13) * 27 + (pw - qw + 13);
        }
        const float s = sr[m] * scale + s_rel[idx];
        sr[m] = s;
        mx = fmaxf(mx, s);
      }
      float sum = 0.f;
      for (int m = 0; m < 197; ++m) {
        const float e = __expf(sr[m] - mx);
        sr[m] = e;
        sum += e;
      }
      const float inv = 1.f / sum;
      for (int m = 0; m < 197; ++m) sr[m] *= inv;
      for (int m = 197; m < NPAD; ++m) sr[m] = 0.f;  // dead key columns
    } else {
      for (int m = 0; m < NPAD; ++m) sr[m] = 0.f;    // dead query rows
    }
  }
  __syncthreads();

  // ---- phase 3: ctx = P @ V   (14 row-tiles x 4 d-tiles, K-dim = 224) ----
  for (int t = wid; t < 56; t += 8) {
    const int ti = t >> 2, dj = t & 3;
    const int m0 = ti * 16, d0 = dj * 16;
    const float*    ap = s_sc + (m0 + (lane & 15)) * NPAD + koffA;
    const _Float16* bp = s_vt + (d0 + (lane & 15)) * NPAD + koffB;
    v8f acc = {};
#pragma unroll
    for (int kc = 0; kc < NPAD; kc += 32) {
      v16h a  = fragA_from_f32(ap + kc);   // f32 probs -> f16 fragment
      v16h bf = *(const v16h*)(bp + kc);
      acc = WMMA_F16(a, bf, acc);
    }
#pragma unroll
    for (int r = 0; r < 8; ++r) {
      const int row = m0 + r + r8;
      if (row < 197)
        ctx[((size_t)b * 197 + row) * 768 + hoff + d0 + (lane & 15)] =
            (_Float16)acc[r];
    }
  }
}

// ---------------------------------------------------------------------------
// Kernel 3: out(12608, 768) f32 = ctx(12608,768) f16 @ proj_wh(768,768)^T + b
// ---------------------------------------------------------------------------
__global__ __launch_bounds__(256) void proj_gemm_kernel(
    const _Float16* __restrict__ A, const _Float16* __restrict__ Wh,
    const float* __restrict__ bias, float* __restrict__ out) {
  constexpr int K = 768, N = 768, NT = 12;  // N / 64
  const int lane = threadIdx.x & 31;
  const int wv = blockIdx.x * 8 + (threadIdx.x >> 5);
  const int mt = wv / NT, nt = wv - mt * NT;
  const int m0 = mt * 32, n0 = nt * 64;

  const _Float16* ap0 = A + (size_t)(m0 + (lane & 15)) * K + ((lane >> 4) << 4);
  const _Float16* ap1 = ap0 + (size_t)16 * K;
  const _Float16* bp  = Wh + (size_t)(n0 + (lane & 15)) * K + ((lane >> 4) << 4);

  v8f acc[2][4] = {};
  for (int kc = 0; kc < K; kc += 32) {
    __builtin_prefetch(ap0 + kc + 256, 0, 1);
    __builtin_prefetch(bp + kc + 256, 0, 1);
    v16h a0 = fragA_from_h(ap0 + kc);
    v16h a1 = fragA_from_h(ap1 + kc);
#pragma unroll
    for (int s = 0; s < 4; ++s) {
      v16h b = *(const v16h*)(bp + (size_t)s * 16 * K + kc);
      acc[0][s] = WMMA_F16(a0, b, acc[0][s]);
      acc[1][s] = WMMA_F16(a1, b, acc[1][s]);
    }
  }

  const int r8 = (lane >> 4) << 3;
#pragma unroll
  for (int s = 0; s < 4; ++s) {
    const int col = n0 + s * 16 + (lane & 15);
    const float bv = bias[col];
#pragma unroll
    for (int half = 0; half < 2; ++half) {
#pragma unroll
      for (int r = 0; r < 8; ++r) {
        const int row = m0 + half * 16 + r + r8;
        out[(size_t)row * N + col] = acc[half][s][r] + bv;
      }
    }
  }
}

// ---------------------------------------------------------------------------
extern "C" void kernel_launch(void* const* d_in, const int* in_sizes, int n_in,
                              void* d_out, int out_size, void* d_ws,
                              size_t ws_size, hipStream_t stream) {
  (void)in_sizes; (void)n_in; (void)out_size; (void)ws_size;
  const float* x         = (const float*)d_in[0];  // (64,197,768)
  const float* qkv_w     = (const float*)d_in[1];  // (2304,768)
  const float* q_bias    = (const float*)d_in[2];
  const float* k_bias    = (const float*)d_in[3];
  const float* v_bias    = (const float*)d_in[4];
  const float* rel_table = (const float*)d_in[5];  // (732,12)
  const float* proj_w    = (const float*)d_in[6];  // (768,768)
  const float* proj_b    = (const float*)d_in[7];
  float* out = (float*)d_out;

  constexpr size_t M = 12608;
  _Float16* xh   = (_Float16*)d_ws;                 // 12608 x 768
  _Float16* wh   = xh + M * 768;                    // 2304 x 768
  _Float16* pwh  = wh + (size_t)2304 * 768;         // 768 x 768
  _Float16* qkvh = pwh + (size_t)768 * 768;         // 12608 x 2304
  _Float16* ctx  = qkvh + M * 2304;                 // 12608 x 768

  cvt_f16_kernel<<<(int)(M * 768 / 1024), 256, 0, stream>>>(x, xh, (int)(M * 768));
  cvt_f16_kernel<<<2304 * 768 / 1024, 256, 0, stream>>>(qkv_w, wh, 2304 * 768);
  cvt_f16_kernel<<<768 * 768 / 1024, 256, 0, stream>>>(proj_w, pwh, 768 * 768);

  // M tiles (32 rows) = 394
  qkv_gemm_kernel<<<(394 * 36) / 8, 256, 0, stream>>>(xh, wh, q_bias, k_bias,
                                                      v_bias, qkvh);

  const size_t lds_bytes =
      (size_t)NPAD * NPAD * 4 + 3 * (size_t)NPAD * 64 * 2 + 732 * 4;  // 289,648
  attn_kernel<<<64 * 12, 256, lds_bytes, stream>>>(qkvh, rel_table, ctx);

  proj_gemm_kernel<<<(394 * 12) / 8, 256, 0, stream>>>(ctx, pwh, proj_b, out);
}
